// InteractionAttention_3968549782227
// MI455X (gfx1250) — matrix operand order repaired and checked by reading, not compiled, not run on hardware
//
#include <hip/hip_runtime.h>

typedef __attribute__((ext_vector_type(16))) __bf16 v16bf;
typedef __attribute__((ext_vector_type(8)))  __bf16 v8bf;
typedef __attribute__((ext_vector_type(8)))  float  v8f;
typedef __attribute__((ext_vector_type(4)))  float  f4;

#define N_   32768
#define K_   4096
#define R_   256
#define ROWS 32            // H rows per block in the fused kernel
#define SCP  (K_ + 8)      // padded LDS score-row stride (bf16 elements)

// ---------------------------------------------------------------------------
// Kernel 1: Q = W_I @ L^T (f32 accumulate), written as bf16 directly in WMMA
// B-fragment order: qfrag[kt][s][lane][e], e <-> k_local = 8*(lane>>4)+(e&7)+16*(e>>3),
// column n = lane&15.  One block per key-tile kt, wave = k-step s.
// ---------------------------------------------------------------------------
__global__ __launch_bounds__(256) void qfrag_kernel(const float* __restrict__ W,
                                                    const float* __restrict__ L,
                                                    __bf16* __restrict__ qfrag) {
  const int kt   = blockIdx.x;           // 0..255 key tiles
  const int s    = threadIdx.x >> 5;     // 0..7 k-steps (contraction over R)
  const int lane = threadIdx.x & 31;
  const int g    = lane >> 4;
  const int col  = kt * 16 + (lane & 15);        // key index (B column)
  const float* lrow = L + (size_t)col * R_;
  float acc[16];
#pragma unroll
  for (int e = 0; e < 16; ++e) acc[e] = 0.f;
  for (int c = 0; c < R_ / 4; ++c) {
    f4 lv = *(const f4*)(lrow + c * 4);
#pragma unroll
    for (int e = 0; e < 16; ++e) {
      const int r = 32 * s + 8 * g + (e & 7) + 16 * (e >> 3);         // Q row (contraction index)
      f4 wv = *(const f4*)(W + (size_t)r * R_ + c * 4);
      acc[e] += wv[0] * lv[0] + wv[1] * lv[1] + wv[2] * lv[2] + wv[3] * lv[3];
    }
  }
  v16bf q;
#pragma unroll
  for (int e = 0; e < 16; ++e) q[e] = (__bf16)acc[e];
  *(v16bf*)(qfrag + (((size_t)kt * 8 + s) * 32 + lane) * 16) = q;
}

// ---------------------------------------------------------------------------
// Kernel 2: re-layout L (f32 -> bf16) into WMMA B-fragment order for C = A@L:
// lfrag[ot][s][lane][e] with k_local = 8*(lane>>4)+(e&7)+16*(e>>3), n = lane&15.
// ---------------------------------------------------------------------------
__global__ __launch_bounds__(256) void lfrag_kernel(const float* __restrict__ L,
                                                    __bf16* __restrict__ lfrag) {
  const int gid  = blockIdx.x * 256 + threadIdx.x;
  const int frag = gid >> 5;             // 0..2047 == ot*128 + s
  const int lane = gid & 31;
  const int s    = frag & 127;
  const int ot   = frag >> 7;
  const int g    = lane >> 4;
  const int col  = ot * 16 + (lane & 15);
  v16bf q;
#pragma unroll
  for (int e = 0; e < 16; ++e) {
    const int k = 32 * s + 8 * g + (e & 7) + 16 * (e >> 3);
    q[e] = (__bf16)L[(size_t)k * R_ + col];
  }
  *(v16bf*)(lfrag + (size_t)frag * 512 + lane * 16) = q;
}

// ---------------------------------------------------------------------------
// Fused kernel: per block, 32 H-rows.  Phase2: scores via WMMA bf16 into LDS.
// Phase3: vectorized softmax (unnormalized exp kept in LDS as bf16,
//         normalized f32 A streamed out coalesced).
// Phase4: C = exp @ L via WMMA; 2 output tiles per wave share each A-fragment
//         and K is split across wave pairs (LDS partial-sum reduction),
//         halving phase-4 LDS read traffic.  1/rowsum folded into epilogue.
// ---------------------------------------------------------------------------
__global__ __launch_bounds__(512, 1) void attn_kernel(const float* __restrict__ H,
                                                      const __bf16* __restrict__ qfrag,
                                                      const __bf16* __restrict__ lfrag,
                                                      float* __restrict__ Cout,
                                                      float* __restrict__ Aout) {
  extern __shared__ char smem[];
  __bf16* sc     = (__bf16*)smem;                          // ROWS x SCP scores
  float*  red    = (float*)(smem + (size_t)ROWS * SCP * 2);
  float*  rowmax = red + 512;
  float*  rowinv = rowmax + ROWS;
  v8f*    pbuf   = (v8f*)(rowinv + ROWS);                  // 8*4*32 v8f = 32KB

  const int tid  = threadIdx.x;
  const int wave = tid >> 5;            // 0..15
  const int lane = tid & 31;
  const int g    = lane >> 4;
  const int m    = lane & 15;
  const int kb   = 8 * g;
  const int rowg = blockIdx.x * ROWS;

  // ---- Phase 1: A-fragments of H (f32 -> bf16) for both 16-row tiles ----
  // 16-bit A layout: lanes 0-15 row M=lane, K in {kb..kb+7} U {16+kb..16+kb+7}
  v16bf afrag[2][8];
#pragma unroll
  for (int rt = 0; rt < 2; ++rt)
#pragma unroll
    for (int s = 0; s < 8; ++s) {
      const float* hrow = H + (size_t)(rowg + rt * 16 + m) * R_ + 32 * s;
      float t[16];
      *(f4*)(t + 0)  = *(const f4*)(hrow + kb);
      *(f4*)(t + 4)  = *(const f4*)(hrow + kb + 4);
      *(f4*)(t + 8)  = *(const f4*)(hrow + 16 + kb);
      *(f4*)(t + 12) = *(const f4*)(hrow + 16 + kb + 4);
      v16bf a;
#pragma unroll
      for (int e = 0; e < 16; ++e) a[e] = (__bf16)t[e];
      afrag[rt][s] = a;
    }

  // ---- Phase 2: scores M = H @ Q via v_wmma_f32_16x16x32_bf16 ----
  for (int t = 0; t < 16; ++t) {
    const int kt = wave * 16 + t;       // key tile handled by this wave
    v8f accA = {0.f, 0.f, 0.f, 0.f, 0.f, 0.f, 0.f, 0.f};
    v8f accB = {0.f, 0.f, 0.f, 0.f, 0.f, 0.f, 0.f, 0.f};
    const __bf16* qp = qfrag + (((size_t)kt * 8) * 32 + lane) * 16;
#pragma unroll
    for (int s = 0; s < 8; ++s) {
      v16bf b = *(const v16bf*)(qp + (size_t)s * 512);
      accA = __builtin_amdgcn_wmma_f32_16x16x32_bf16(false, afrag[0][s], false, b,
                                                     (short)0, accA, false, false);
      accB = __builtin_amdgcn_wmma_f32_16x16x32_bf16(false, afrag[1][s], false, b,
                                                     (short)0, accB, false, false);
    }
    if (t + 1 < 16) __builtin_prefetch(qp + 8 * 512, 0, 1);   // next key tile
#pragma unroll
    for (int v = 0; v < 8; ++v) {
      const int mr = v + 8 * g;         // D row within tile
      sc[(size_t)mr        * SCP + kt * 16 + m] = (__bf16)accA[v];
      sc[(size_t)(16 + mr) * SCP + kt * 16 + m] = (__bf16)accB[v];
    }
  }
  __syncthreads();

  // ---- Phase 3: row softmax over K=4096, vectorized 8-wide (v8bf) ----
  {
    const int row = tid >> 4;           // 0..31
    const int sub = tid & 15;
    __bf16* srow = sc + (size_t)row * SCP;
    float lm = -3.402823e38f;
    for (int i = 0; i < 32; ++i) {
      v8bf v = *(const v8bf*)(srow + (i * 16 + sub) * 8);
#pragma unroll
      for (int e = 0; e < 8; ++e) lm = fmaxf(lm, (float)v[e]);
    }
    red[tid] = lm;
    __syncthreads();
    if (sub == 0) {
      float mx = red[tid];
#pragma unroll
      for (int j = 1; j < 16; ++j) mx = fmaxf(mx, red[tid + j]);
      rowmax[row] = mx;
    }
    __syncthreads();
    const float mx = rowmax[row];
    float ls = 0.f;
    for (int i = 0; i < 32; ++i) {
      const int off = (i * 16 + sub) * 8;
      v8bf v = *(const v8bf*)(srow + off);
      v8bf o;
#pragma unroll
      for (int e = 0; e < 8; ++e) {
        const float p = __expf((float)v[e] - mx);
        o[e] = (__bf16)p;
        ls += p;
      }
      *(v8bf*)(srow + off) = o;         // keep unnormalized exp for phase 4
    }
    red[tid] = ls;
    __syncthreads();
    if (sub == 0) {
      float sm = red[tid];
#pragma unroll
      for (int j = 1; j < 16; ++j) sm += red[tid + j];
      rowinv[row] = 1.f / sm;
    }
    __syncthreads();

    // normalized A_I (f32) store: per thread 32B chunks, 512B coalesced runs
    const float inv = rowinv[row];
    float* arow = Aout + (size_t)(rowg + row) * K_;
    for (int i = 0; i < 32; ++i) {
      const int off = (i * 16 + sub) * 8;
      v8bf v = *(const v8bf*)(srow + off);
      f4 o0, o1;
#pragma unroll
      for (int e = 0; e < 4; ++e) { o0[e] = (float)v[e] * inv; o1[e] = (float)v[e + 4] * inv; }
      *(f4*)(arow + off)     = o0;
      *(f4*)(arow + off + 4) = o1;
    }
  }

  // ---- Phase 4: C = exp @ L via WMMA.  wave = (otp 0..7, kh 0..1):
  // 2 output tiles per wave share each A-fragment; K split across kh. ----
  {
    const int otp = wave & 7;
    const int kh  = wave >> 3;
    const int ot0 = otp * 2;
    const int ot1 = otp * 2 + 1;
    v8f acc[4];                          // (ot0,rt0) (ot0,rt1) (ot1,rt0) (ot1,rt1)
#pragma unroll
    for (int i = 0; i < 4; ++i) acc[i] = (v8f){0.f,0.f,0.f,0.f,0.f,0.f,0.f,0.f};
    const __bf16* lp0 = lfrag + ((size_t)ot0 * 128 * 32 + lane) * 16;
    const __bf16* lp1 = lfrag + ((size_t)ot1 * 128 * 32 + lane) * 16;
    const int s0 = kh * 64;
    for (int s = s0; s < s0 + 64; ++s) {
      const __bf16* pa = sc + (size_t)m * SCP + 32 * s;           // row tile 0
      const __bf16* pb = sc + (size_t)(16 + m) * SCP + 32 * s;    // row tile 1
      v8bf a0 = *(const v8bf*)(pa + kb);          // K = kb..kb+7      (16B LDS)
      v8bf a1 = *(const v8bf*)(pa + 16 + kb);     // K = 16+kb..16+kb+7
      v8bf b0 = *(const v8bf*)(pb + kb);
      v8bf b1 = *(const v8bf*)(pb + 16 + kb);
      v16bf aA, aB;
#pragma unroll
      for (int e = 0; e < 8; ++e) {
        aA[e] = a0[e]; aA[e + 8] = a1[e];
        aB[e] = b0[e]; aB[e + 8] = b1[e];
      }
      v16bf bf0 = *(const v16bf*)(lp0 + (size_t)s * 512);
      v16bf bf1 = *(const v16bf*)(lp1 + (size_t)s * 512);
      acc[0] = __builtin_amdgcn_wmma_f32_16x16x32_bf16(false, aA, false, bf0,
                                                       (short)0, acc[0], false, false);
      acc[1] = __builtin_amdgcn_wmma_f32_16x16x32_bf16(false, aB, false, bf0,
                                                       (short)0, acc[1], false, false);
      acc[2] = __builtin_amdgcn_wmma_f32_16x16x32_bf16(false, aA, false, bf1,
                                                       (short)0, acc[2], false, false);
      acc[3] = __builtin_amdgcn_wmma_f32_16x16x32_bf16(false, aB, false, bf1,
                                                       (short)0, acc[3], false, false);
    }

    // split-K reduction: kh==1 waves publish partials, kh==0 waves combine
    if (kh == 1) {
#pragma unroll
      for (int ti = 0; ti < 4; ++ti)
        pbuf[((size_t)otp * 4 + ti) * 32 + lane] = acc[ti];
    }
    __syncthreads();
    if (kh == 0) {
#pragma unroll
      for (int ti = 0; ti < 4; ++ti) {
        v8f part = pbuf[((size_t)otp * 4 + ti) * 32 + lane];
        const int rtb = (ti & 1) * 16;                 // row-tile base
        const int oc  = (ot0 + (ti >> 1)) * 16 + m;    // output column
#pragma unroll
        for (int v = 0; v < 8; ++v) {
          const int mr = rtb + v + 8 * g;
          Cout[(size_t)(rowg + mr) * R_ + oc] = (acc[ti][v] + part[v]) * rowinv[mr];
        }
      }
    }
  }
}

// ---------------------------------------------------------------------------
extern "C" void kernel_launch(void* const* d_in, const int* in_sizes, int n_in,
                              void* d_out, int out_size, void* d_ws, size_t ws_size,
                              hipStream_t stream) {
  (void)in_sizes; (void)n_in; (void)out_size; (void)ws_size;
  const float* H = (const float*)d_in[0];   // (N, R)
  const float* L = (const float*)d_in[1];   // (K, R)
  const float* W = (const float*)d_in[2];   // (R, R)

  float* Cout = (float*)d_out;                      // (N, R) first
  float* Aout = Cout + (size_t)N_ * R_;             // then (N, K)

  __bf16* qfrag = (__bf16*)d_ws;                    // 256*8*32*16 = 1M bf16 (2MB)
  __bf16* lfrag = qfrag + (size_t)256 * 8 * 512;    // 16*128*32*16 = 1M bf16 (2MB)

  qfrag_kernel<<<K_ / 16, 256, 0, stream>>>(W, L, qfrag);
  lfrag_kernel<<<256, 256, 0, stream>>>(L, lfrag);

  const size_t smem = (size_t)ROWS * SCP * 2   // bf16 score buffer (262,656B)
                    + 512 * sizeof(float)      // per-thread reduction
                    + 2 * ROWS * sizeof(float) // rowmax + rowinv
                    + 8 * 4 * 32 * sizeof(v8f);// split-K partial buffer (32KB)
  attn_kernel<<<N_ / ROWS, 512, smem, stream>>>(H, qfrag, lfrag, Cout, Aout);
}
